// GraphSAGE_31585189495318
// MI455X (gfx1250) — compile-verified, matrix operand order checked by reading, not run on hardware
//
#include <hip/hip_runtime.h>
#include <math.h>
#include <stdint.h>

// GraphSAGE (2-layer, mean aggregation) for MI455X / gfx1250.
//
// Roofline: edge aggregation dominates; it is atomic-op-count bound, so we
// accumulate agg in f16 with packed GLOBAL_ATOMIC_PK_ADD_F16 (halves atomic
// ops + bytes; agg table is 12.8MB -> L2-resident at 192MB L2).
// The 1/cnt mean-scale commutes with the GEMM, so A-tiles are pure f16 byte
// copies -> staged into LDS with CDNA5 async-DMA (global_load_async_to_lds_b128
// + s_wait_asynccnt). GEMMs use v_wmma_f32_16x16x32_f16 (f16 A/B, f32 acc).

typedef __attribute__((ext_vector_type(16))) _Float16 v16h;
typedef __attribute__((ext_vector_type(8)))  _Float16 v8h;
typedef __attribute__((ext_vector_type(8)))  float    v8f;

#define N_NODES 50000
#define N_EDGES 600000
#define DIMK    128     // K of both layers (IN_DIM == HID == 128)

// ---------------------------------------------------------------- utilities

__global__ void zero_b32_kernel(uint32_t* __restrict__ p, int n) {
  for (int i = blockIdx.x * blockDim.x + threadIdx.x; i < n;
       i += gridDim.x * blockDim.x)
    p[i] = 0u;
}

__global__ void f32_to_f16_kernel(const float* __restrict__ in,
                                  _Float16* __restrict__ out, int n) {
  int i = blockIdx.x * blockDim.x + threadIdx.x;
  if (i < n) out[i] = (_Float16)in[i];
}

// in-degree counts (float so the gemm kernel can fmaxf(cnt,1))
__global__ void degree_count_kernel(const long long* __restrict__ dst,
                                    float* __restrict__ cnt, int ne) {
  int e = blockIdx.x * blockDim.x + threadIdx.x;
  if (e < ne) atomicAdd(&cnt[(int)dst[e]], 1.0f);
}

// one warp per edge: lane handles 2 packed-f16 pairs of the 128 features.
// Gathers hit L2 (f16 tables are L2-resident); scatter via
// global_atomic_pk_add_f16 (2 halves per atomic -> half the atomic ops).
__global__ void scatter_pk_add_kernel(const _Float16* __restrict__ feat,
                                      const long long* __restrict__ src,
                                      const long long* __restrict__ dst,
                                      _Float16* __restrict__ agg, int ne) {
  int warp = (blockIdx.x * blockDim.x + threadIdx.x) >> 5;
  int lane = threadIdx.x & 31;
  if (warp >= ne) return;
  int s = (int)src[warp];
  int d = (int)dst[warp];
  const uint32_t* fs = (const uint32_t*)(feat + (size_t)s * DIMK);  // 64 half2
  uint64_t ad = (uint64_t)(uintptr_t)(agg + (size_t)d * DIMK);
  __builtin_prefetch(fs, 0, 0);  // global_prefetch_b8
#pragma unroll
  for (int j = 0; j < 2; ++j) {
    int i = lane + 32 * j;                 // half2 index 0..63
    uint32_t v = fs[i];
    uint64_t a = ad + (uint64_t)i * 4u;
    asm volatile("global_atomic_pk_add_f16 %0, %1, off"
                 :: "v"(a), "v"(v) : "memory");
  }
}

// ------------------------------------------------- fused SAGE layer (WMMA)
//
// out[16 x NOUT] block per workgroup:
//   h = (agg@Wl^T)/cnt + self@Wr^T + bias ; h = l2norm_row(sigmoid(h))
// (1/cnt folded into the epilogue -> phase 0 is a pure f16 tile copy, done
//  with async-DMA to LDS.)
// One wave per 16-col tile; K=128 consumed as 4 x (16x16x32) WMMA per matmul,
// separate f32 accumulators for the agg and self matmuls.
//
// VGPR layouts per ISA 7.12.2 (wave32):
//   A: lane l -> row l&15 ; halves 0..7  = K[kb+hih .. +8)
//                           halves 8..15 = K[kb+16+hih .. +8), hih = 8*(l>=16)
//   B: same with col l&15 (weights stored [NOUT,K] row-major == B^T rows)
//   D: lane l, vgpr j -> (row j + 8*(l>=16), col l&15)

template <int NOUT, int NWAVES, typename OutT>
__global__ __launch_bounds__(NWAVES * 32)
void sage_wmma_kernel(const _Float16* __restrict__ aggH,   // f16 agg (unscaled)
                      const float* __restrict__ cnt,
                      const _Float16* __restrict__ selfH,  // f16 self features
                      const _Float16* __restrict__ Wl,
                      const _Float16* __restrict__ Wr,
                      const float* __restrict__ bias,
                      OutT* __restrict__ out) {
  constexpr int K = DIMK;
  __shared__ __align__(16) _Float16 ldsA[16 * K];  // agg tile, f16 (4KB)
  __shared__ __align__(16) _Float16 ldsS[16 * K];  // self tile, f16 (4KB)
  __shared__ float rcs[16];    // 1/max(cnt,1) per row
  __shared__ float rowsq[16];
  __shared__ float rinv[16];

  const int tid  = (int)threadIdx.x;
  const int wave = tid >> 5;
  const int lane = tid & 31;
  const int row0 = (int)blockIdx.x * 16;  // node base (50000 = 3125*16)

  if (tid < 16) {
    rcs[tid]   = 1.0f / fmaxf(cnt[row0 + tid], 1.0f);
    rowsq[tid] = 0.0f;
  }

  // phase 0: async-DMA both 4KB f16 tiles into LDS (ASYNCcnt-tracked).
  // Generic pointers to LDS carry the LDS byte offset in their low 32 bits.
  {
    const uint64_t gA = (uint64_t)(uintptr_t)(aggH  + (size_t)row0 * K);
    const uint64_t gS = (uint64_t)(uintptr_t)(selfH + (size_t)row0 * K);
    const uint32_t lA = (uint32_t)(uintptr_t)&ldsA[0];
    const uint32_t lS = (uint32_t)(uintptr_t)&ldsS[0];
    constexpr int TILE_BYTES = 16 * K * (int)sizeof(_Float16);  // 4096
    for (int ofs = tid * 16; ofs < TILE_BYTES; ofs += NWAVES * 32 * 16) {
      uint32_t la = lA + (uint32_t)ofs;
      uint64_t ga = gA + (uint64_t)ofs;
      asm volatile("global_load_async_to_lds_b128 %0, %1, off"
                   :: "v"(la), "v"(ga) : "memory");
      uint32_t ls = lS + (uint32_t)ofs;
      uint64_t gs = gS + (uint64_t)ofs;
      asm volatile("global_load_async_to_lds_b128 %0, %1, off"
                   :: "v"(ls), "v"(gs) : "memory");
    }
    asm volatile("s_wait_asynccnt 0" ::: "memory");
  }
  __syncthreads();

  // phase 1: WMMA over K (uniform control flow, EXEC all-1s)
  const int l15 = lane & 15;
  const int hih = (lane >> 4) * 8;  // 8-half offset for hi half-wave
  const int n0  = wave * 16;
  const _Float16* wl = Wl + (size_t)(n0 + l15) * K;
  const _Float16* wr = Wr + (size_t)(n0 + l15) * K;

  v8f accA = {};
  v8f accS = {};
#pragma unroll
  for (int kb = 0; kb < K; kb += 32) {
    v8h a0 = *(const v8h*)&ldsA[l15 * K + kb + hih];
    v8h a1 = *(const v8h*)&ldsA[l15 * K + kb + 16 + hih];
    v16h a = __builtin_shufflevector(a0, a1, 0, 1, 2, 3, 4, 5, 6, 7,
                                     8, 9, 10, 11, 12, 13, 14, 15);
    v8h b0 = *(const v8h*)(wl + kb + hih);
    v8h b1 = *(const v8h*)(wl + kb + 16 + hih);
    v16h b = __builtin_shufflevector(b0, b1, 0, 1, 2, 3, 4, 5, 6, 7,
                                     8, 9, 10, 11, 12, 13, 14, 15);
    accA = __builtin_amdgcn_wmma_f32_16x16x32_f16(false, a, false, b,
                                                  (short)0, accA, false, false);

    v8h s0 = *(const v8h*)&ldsS[l15 * K + kb + hih];
    v8h s1 = *(const v8h*)&ldsS[l15 * K + kb + 16 + hih];
    v16h sa = __builtin_shufflevector(s0, s1, 0, 1, 2, 3, 4, 5, 6, 7,
                                      8, 9, 10, 11, 12, 13, 14, 15);
    v8h r0 = *(const v8h*)(wr + kb + hih);
    v8h r1 = *(const v8h*)(wr + kb + 16 + hih);
    v16h rb = __builtin_shufflevector(r0, r1, 0, 1, 2, 3, 4, 5, 6, 7,
                                      8, 9, 10, 11, 12, 13, 14, 15);
    accS = __builtin_amdgcn_wmma_f32_16x16x32_f16(false, sa, false, rb,
                                                  (short)0, accS, false, false);
  }

  // epilogue: mean-scale + bias + sigmoid, then row-wise L2 norm
  const int rbase = (lane >> 4) * 8;
  const float bn  = bias[n0 + l15];
  float vals[8];
#pragma unroll
  for (int j = 0; j < 8; ++j) {
    int r = j + rbase;
    float v = accA[j] * rcs[r] + accS[j] + bn;
    v = 1.0f / (1.0f + __expf(-v));   // sigmoid
    vals[j] = v;
    atomicAdd(&rowsq[r], v * v);      // ds_add_f32
  }
  __syncthreads();
  if (tid < 16) rinv[tid] = 1.0f / fmaxf(sqrtf(rowsq[tid]), 1e-12f);
  __syncthreads();
#pragma unroll
  for (int j = 0; j < 8; ++j) {
    int r = j + rbase;
    out[(size_t)(row0 + r) * NOUT + n0 + l15] = (OutT)(vals[j] * rinv[r]);
  }
}

// ----------------------------------------------------------------- launch

static inline size_t align_up(size_t x, size_t a) { return (x + a - 1) & ~(a - 1); }

extern "C" void kernel_launch(void* const* d_in, const int* in_sizes, int n_in,
                              void* d_out, int out_size, void* d_ws, size_t ws_size,
                              hipStream_t stream) {
  // inputs (setup_inputs order)
  const float*     x    = (const float*)d_in[0];                 // [50000,128]
  const long long* ei   = (const long long*)d_in[1];             // [2,600000] int64
  const float*     W1l  = (const float*)d_in[2];                 // [128,128]
  const float*     W1r  = (const float*)d_in[3];                 // [128,128]
  const float*     b1   = (const float*)d_in[4];                 // [128]
  const float*     W2l  = (const float*)d_in[5];                 // [256,128]
  const float*     W2r  = (const float*)d_in[6];                 // [256,128]
  const float*     b2   = (const float*)d_in[7];                 // [256]
  float*           out  = (float*)d_out;                         // [50000,256]

  const long long* src = ei;            // edge_index[0]
  const long long* dst = ei + N_EDGES;  // edge_index[1]

  // workspace carve-out (256B aligned)
  char* ws = (char*)d_ws;
  size_t off = 0;
  float*    cnt   = (float*)(ws + off);    off = align_up(off + (size_t)N_NODES * 4, 256);
  _Float16* aggH  = (_Float16*)(ws + off); off = align_up(off + (size_t)N_NODES * DIMK * 2, 256);
  _Float16* xH    = (_Float16*)(ws + off); off = align_up(off + (size_t)N_NODES * DIMK * 2, 256);
  _Float16* h1H   = (_Float16*)(ws + off); off = align_up(off + (size_t)N_NODES * DIMK * 2, 256);
  _Float16* W1l_h = (_Float16*)(ws + off); off = align_up(off + (size_t)128 * 128 * 2, 256);
  _Float16* W1r_h = (_Float16*)(ws + off); off = align_up(off + (size_t)128 * 128 * 2, 256);
  _Float16* W2l_h = (_Float16*)(ws + off); off = align_up(off + (size_t)256 * 128 * 2, 256);
  _Float16* W2r_h = (_Float16*)(ws + off); off = align_up(off + (size_t)256 * 128 * 2, 256);
  (void)ws_size; (void)n_in; (void)in_sizes; (void)out_size;

  // weights + x -> f16
  f32_to_f16_kernel<<<(128 * 128 + 255) / 256, 256, 0, stream>>>(W1l, W1l_h, 128 * 128);
  f32_to_f16_kernel<<<(128 * 128 + 255) / 256, 256, 0, stream>>>(W1r, W1r_h, 128 * 128);
  f32_to_f16_kernel<<<(256 * 128 + 255) / 256, 256, 0, stream>>>(W2l, W2l_h, 256 * 128);
  f32_to_f16_kernel<<<(256 * 128 + 255) / 256, 256, 0, stream>>>(W2r, W2r_h, 256 * 128);
  f32_to_f16_kernel<<<(N_NODES * DIMK + 255) / 256, 256, 0, stream>>>(
      x, xH, N_NODES * DIMK);

  // degree counts
  zero_b32_kernel<<<(N_NODES + 255) / 256, 256, 0, stream>>>((uint32_t*)cnt, N_NODES);
  degree_count_kernel<<<(N_EDGES + 255) / 256, 256, 0, stream>>>(dst, cnt, N_EDGES);

  const int aggDwords = N_NODES * DIMK / 2;  // f16 agg as dwords

  // ---- layer 1 ----
  zero_b32_kernel<<<2048, 256, 0, stream>>>((uint32_t*)aggH, aggDwords);
  scatter_pk_add_kernel<<<N_EDGES / 8, 256, 0, stream>>>(xH, src, dst, aggH, N_EDGES);
  sage_wmma_kernel<128, 8, _Float16><<<N_NODES / 16, 8 * 32, 0, stream>>>(
      aggH, cnt, xH, W1l_h, W1r_h, b1, h1H);

  // ---- layer 2 ----
  zero_b32_kernel<<<2048, 256, 0, stream>>>((uint32_t*)aggH, aggDwords);
  scatter_pk_add_kernel<<<N_EDGES / 8, 256, 0, stream>>>(h1H, src, dst, aggH, N_EDGES);
  sage_wmma_kernel<256, 16, float><<<N_NODES / 16, 16 * 32, 0, stream>>>(
      aggH, cnt, h1H, W2l_h, W2r_h, b2, out);
}
